// Encoder_Decoder_60576218743336
// MI455X (gfx1250) — matrix operand order repaired and checked by reading, not compiled, not run on hardware
//
#include <hip/hip_runtime.h>
#include <math.h>

#define NHEADS 16
#define EDIM   1024
#define FFDIM  4096
#define HD     64
#define NLAYER 4
#define VOCT   32000
#define BSZ    4
#define SSRC   512
#define STGT   256
#define ATT_SCALE 0.125f
#define LNEPS  1e-6f

typedef __attribute__((ext_vector_type(16))) _Float16 v16h;
typedef __attribute__((ext_vector_type(8)))  _Float16 v8h;
typedef __attribute__((ext_vector_type(4)))  _Float16 v4h;
typedef __attribute__((ext_vector_type(8)))  float    v8f;

__device__ __forceinline__ v16h cat16(v8h lo, v8h hi) {
    return __builtin_shufflevector(lo, hi, 0,1,2,3,4,5,6,7,8,9,10,11,12,13,14,15);
}

// ---------------------------------------------------------------------------
// Generic GEMM: out = act(A[MxK] @ W[KxN] + bias (+ res)), f16 WMMA, f32 accum
// flags bit0 = relu, bit1 = head-major scatter store (QKV projections)
// LDS layouts chosen so every WMMA fragment is contiguous 16B-aligned runs:
//   As [m][k]  stride 40 halfs (80B)  -> A-frag = 2x ds_load_b128
//   BsT[n][k]  stride 40 halfs (80B)  -> B-frag = 2x ds_load_b128
// ---------------------------------------------------------------------------
__global__ __launch_bounds__(256) void gemm_f16wmma(
    const float* __restrict__ A, const float* __restrict__ W,
    const float* __restrict__ bias, const float* __restrict__ res,
    float* __restrict__ out, int M, int N, int K, int flags, int sl)
{
    __shared__ __align__(16) _Float16 As [128][40];
    __shared__ __align__(16) _Float16 BsT[128][40];

    const int tid  = threadIdx.x;
    const int lane = tid & 31;
    const int wave = tid >> 5;
    const int half = lane >> 4;
    const int nl   = lane & 15;
    const long m_base = (long)blockIdx.y * 128;
    const long n_base = (long)blockIdx.x * 128;

    v8f acc[8];
#pragma unroll
    for (int t = 0; t < 8; ++t) acc[t] = {};

    // A staging: thread -> (row ar, 16 consecutive k), fully coalesced float4
    const int ar = tid >> 1;
    const int ac = (tid & 1) * 16;
    // B staging: thread -> 4k x 4n block; float4-coalesced reads, b64 stores
    const int kb = (tid >> 5) * 4;
    const int nb = (tid & 31) * 4;

    for (int k0 = 0; k0 < K; k0 += 32) {
        {   // ---- stage A tile (128 x 32) ----
            const float* ag = A + (m_base + ar) * (long)K + k0 + ac;
            const float4 p0 = *(const float4*)(ag + 0);
            const float4 p1 = *(const float4*)(ag + 4);
            const float4 p2 = *(const float4*)(ag + 8);
            const float4 p3 = *(const float4*)(ag + 12);
            v8h lo = {(_Float16)p0.x,(_Float16)p0.y,(_Float16)p0.z,(_Float16)p0.w,
                      (_Float16)p1.x,(_Float16)p1.y,(_Float16)p1.z,(_Float16)p1.w};
            v8h hi = {(_Float16)p2.x,(_Float16)p2.y,(_Float16)p2.z,(_Float16)p2.w,
                      (_Float16)p3.x,(_Float16)p3.y,(_Float16)p3.z,(_Float16)p3.w};
            *(v8h*)&As[ar][ac]     = lo;
            *(v8h*)&As[ar][ac + 8] = hi;
        }
        {   // ---- stage B tile transposed (BsT[n][k]) ----
            const float* bg = W + (long)(k0 + kb) * N + n_base + nb;
            const float4 w0 = *(const float4*)(bg + 0L * N);
            const float4 w1 = *(const float4*)(bg + 1L * N);
            const float4 w2 = *(const float4*)(bg + 2L * N);
            const float4 w3 = *(const float4*)(bg + 3L * N);
            v4h c0 = {(_Float16)w0.x,(_Float16)w1.x,(_Float16)w2.x,(_Float16)w3.x};
            v4h c1 = {(_Float16)w0.y,(_Float16)w1.y,(_Float16)w2.y,(_Float16)w3.y};
            v4h c2 = {(_Float16)w0.z,(_Float16)w1.z,(_Float16)w2.z,(_Float16)w3.z};
            v4h c3 = {(_Float16)w0.w,(_Float16)w1.w,(_Float16)w2.w,(_Float16)w3.w};
            *(v4h*)&BsT[nb + 0][kb] = c0;
            *(v4h*)&BsT[nb + 1][kb] = c1;
            *(v4h*)&BsT[nb + 2][kb] = c2;
            *(v4h*)&BsT[nb + 3][kb] = c3;
        }
        __syncthreads();

        // A fragment: rows = wave's 16 M-rows; two aligned 16B runs
        const _Float16* arow = &As[wave * 16 + nl][0];
        const v16h af = cat16(*(const v8h*)(arow + 8 * half),
                              *(const v8h*)(arow + 16 + 8 * half));
#pragma unroll
        for (int t = 0; t < 8; ++t) {
            const _Float16* brow = &BsT[t * 16 + nl][0];
            const v16h bf = cat16(*(const v8h*)(brow + 8 * half),
                                  *(const v8h*)(brow + 16 + 8 * half));
            acc[t] = __builtin_amdgcn_wmma_f32_16x16x32_f16(
                false, af, false, bf, (short)0, acc[t], false, false);
        }
        __syncthreads();
    }

    // epilogue (tile never crosses a batch boundary: sl % 128 == 0)
    const long bb = (flags & 2) ? (m_base / sl) : 0;
    const long tb = (flags & 2) ? (m_base - bb * sl) : 0;
#pragma unroll
    for (int t = 0; t < 8; ++t) {
#pragma unroll
        for (int r = 0; r < 8; ++r) {
            const long mr = wave * 16 + r + 8 * half;
            const long gm = m_base + mr;
            const long gn = n_base + t * 16 + nl;
            float v = acc[t][r] + bias[gn];
            if (res) v += res[gm * (long)N + gn];
            if (flags & 1) v = fmaxf(v, 0.0f);
            if (flags & 2) {
                const long tt = tb + mr;
                const long d = gn >> 4, h = gn & 15;
                out[(((bb * NHEADS + h) * sl + tt) << 6) + d] = v;
            } else {
                out[gm * (long)N + gn] = v;
            }
        }
    }
}

// ---------------------------------------------------------------------------
// Flash attention, one wave per (b, h, 16-q tile).
// Q/K/V head-major [b,h,t,64] fp32 -> output [b,q,h*64+d].
// S^T = K*Q^T (softmax over k = per-lane + one shfl_xor); Z^T = V^T * P^T.
// LDS: Ql[q][d] s72, Kl[k][d] s72, Vt[d][k] s40, Plt[q][k] s40 — all frags
// are contiguous aligned runs (ds_load_b128) and P is stored as b128.
// ---------------------------------------------------------------------------
__global__ __launch_bounds__(32) void attn_flash(
    const float* __restrict__ Qh, const float* __restrict__ Kh,
    const float* __restrict__ Vh, const int* __restrict__ tok,
    float* __restrict__ out, int slq, int slk, int causal)
{
    __shared__ __align__(16) _Float16 Ql [16][72];
    __shared__ __align__(16) _Float16 Kl [32][72];
    __shared__ __align__(16) _Float16 Vt [64][40];
    __shared__ __align__(16) _Float16 Plt[16][40];

    const int lane = threadIdx.x;
    const int half = lane >> 4;
    const int nl   = lane & 15;
    const int q0   = blockIdx.x * 16;
    const int h    = blockIdx.y;
    const int b    = blockIdx.z;
    const int qg   = q0 + nl;

    const float* Qb = Qh + (((long)b * NHEADS + h) * slq) * HD;
    const float* Kb = Kh + (((long)b * NHEADS + h) * slk) * HD;
    const float* Vb = Vh + (((long)b * NHEADS + h) * slk) * HD;

    // stage Q tile (16 x 64), float4-coalesced, packed b64 stores
    for (int i = lane; i < 16 * 16; i += 32) {
        const int r = i >> 4, c = (i & 15) * 4;
        const float4 p = *(const float4*)&Qb[(q0 + r) * HD + c];
        v4h q4 = {(_Float16)p.x,(_Float16)p.y,(_Float16)p.z,(_Float16)p.w};
        *(v4h*)&Ql[r][c] = q4;
    }

    v8f zt[4];
#pragma unroll
    for (int t = 0; t < 4; ++t) zt[t] = {};
    float m_run = -1e30f, l_run = 0.0f;

    for (int k0 = 0; k0 < slk; k0 += 32) {
        __syncthreads();
        // stage K chunk (32 x 64)
        for (int i = lane; i < 32 * 16; i += 32) {
            const int r = i >> 4, c = (i & 15) * 4;
            const float4 p = *(const float4*)&Kb[(k0 + r) * HD + c];
            v4h k4 = {(_Float16)p.x,(_Float16)p.y,(_Float16)p.z,(_Float16)p.w};
            *(v4h*)&Kl[r][c] = k4;
        }
        // stage V chunk transposed: Vt[d][k], 4k x 4d blocks
        for (int it = 0; it < 4; ++it) {
            const int blk = it * 32 + lane;        // 0..127
            const int vk  = (blk >> 4) * 4;        // k start
            const int vd  = (blk & 15) * 4;        // d start
            const float* vg = &Vb[(k0 + vk) * HD + vd];
            const float4 w0 = *(const float4*)(vg + 0 * HD);
            const float4 w1 = *(const float4*)(vg + 1 * HD);
            const float4 w2 = *(const float4*)(vg + 2 * HD);
            const float4 w3 = *(const float4*)(vg + 3 * HD);
            v4h c0 = {(_Float16)w0.x,(_Float16)w1.x,(_Float16)w2.x,(_Float16)w3.x};
            v4h c1 = {(_Float16)w0.y,(_Float16)w1.y,(_Float16)w2.y,(_Float16)w3.y};
            v4h c2 = {(_Float16)w0.z,(_Float16)w1.z,(_Float16)w2.z,(_Float16)w3.z};
            v4h c3 = {(_Float16)w0.w,(_Float16)w1.w,(_Float16)w2.w,(_Float16)w3.w};
            *(v4h*)&Vt[vd + 0][vk] = c0;
            *(v4h*)&Vt[vd + 1][vk] = c1;
            *(v4h*)&Vt[vd + 2][vk] = c2;
            *(v4h*)&Vt[vd + 3][vk] = c3;
        }
        __syncthreads();

        // per-chunk validity bitmask: lane tests token k0+lane, ballot
        const bool tokOk = (tok[(long)b * slk + k0 + lane] != 0);
        const unsigned long long bm = __ballot(tokOk);

        float s[2][8];
#pragma unroll
        for (int kt = 0; kt < 2; ++kt) {
            v8f sc = {};
#pragma unroll
            for (int dc = 0; dc < 2; ++dc) {
                const _Float16* krow = &Kl[kt * 16 + nl][0];
                const v16h ak = cat16(*(const v8h*)(krow + dc*32 + 8*half),
                                      *(const v8h*)(krow + dc*32 + 16 + 8*half));
                const _Float16* qrow = &Ql[nl][0];
                const v16h bq = cat16(*(const v8h*)(qrow + dc*32 + 8*half),
                                      *(const v8h*)(qrow + dc*32 + 16 + 8*half));
                sc = __builtin_amdgcn_wmma_f32_16x16x32_f16(
                    false, ak, false, bq, (short)0, sc, false, false);
            }
#pragma unroll
            for (int r = 0; r < 8; ++r) {
                const int kl = kt * 16 + 8 * half + r;   // local k (contiguous in r)
                const int kg = k0 + kl;
                const bool valid = ((bm >> kl) & 1ull) && (!causal || kg <= qg);
                const float sv = valid ? sc[r] : -1e9f;  // mask before scale
                s[kt][r] = sv * ATT_SCALE;
            }
        }
        // online softmax (column q = nl)
        float cm = -1e30f;
#pragma unroll
        for (int kt = 0; kt < 2; ++kt)
#pragma unroll
            for (int r = 0; r < 8; ++r) cm = fmaxf(cm, s[kt][r]);
        cm = fmaxf(cm, __shfl_xor(cm, 16, 32));
        const float m_new = fmaxf(m_run, cm);
        const float corr  = __expf(m_run - m_new);
        float psum = 0.0f;
        float p[2][8];
#pragma unroll
        for (int kt = 0; kt < 2; ++kt)
#pragma unroll
            for (int r = 0; r < 8; ++r) { p[kt][r] = __expf(s[kt][r] - m_new); psum += p[kt][r]; }
        psum += __shfl_xor(psum, 16, 32);
        l_run = l_run * corr + psum;
        m_run = m_new;
#pragma unroll
        for (int kt = 0; kt < 2; ++kt) {     // P^T as one b128 store per kt
            v8h pp = {(_Float16)p[kt][0],(_Float16)p[kt][1],(_Float16)p[kt][2],
                      (_Float16)p[kt][3],(_Float16)p[kt][4],(_Float16)p[kt][5],
                      (_Float16)p[kt][6],(_Float16)p[kt][7]};
            *(v8h*)&Plt[nl][kt * 16 + 8 * half] = pp;
        }
#pragma unroll
        for (int t = 0; t < 4; ++t)
#pragma unroll
            for (int r = 0; r < 8; ++r) zt[t][r] *= corr;
        __syncthreads();

        // Z^T += V^T (16d x 32k) * P^T (32k x 16q)
        const _Float16* prow = &Plt[nl][0];
        const v16h bp = cat16(*(const v8h*)(prow + 8 * half),
                              *(const v8h*)(prow + 16 + 8 * half));
#pragma unroll
        for (int dt = 0; dt < 4; ++dt) {
            const _Float16* vrow = &Vt[dt * 16 + nl][0];
            const v16h av = cat16(*(const v8h*)(vrow + 8 * half),
                                  *(const v8h*)(vrow + 16 + 8 * half));
            zt[dt] = __builtin_amdgcn_wmma_f32_16x16x32_f16(
                false, av, false, bp, (short)0, zt[dt], false, false);
        }
    }

    const float inv = 1.0f / l_run;
#pragma unroll
    for (int dt = 0; dt < 4; ++dt)
#pragma unroll
        for (int r = 0; r < 8; ++r) {
            const int d = dt * 16 + r + 8 * half;
            out[((long)b * slq + (q0 + nl)) * EDIM + h * HD + d] = zt[dt][r] * inv;
        }
}

// ---------------------------------------------------------------------------
// Embedding gather + positional encoding (faithful pe[b] indexing!)
// ---------------------------------------------------------------------------
__global__ __launch_bounds__(256) void embed_pe(
    const int* __restrict__ tok, const float* __restrict__ emb,
    float* __restrict__ out, int sl)
{
    const long t  = blockIdx.x;
    const int  b  = (int)(t / sl);
    const int  tk = tok[t];
    for (int e = threadIdx.x; e < EDIM; e += 256) {
        const int   i2  = e & ~1;
        const float div = __expf(-(float)i2 * (9.210340371976184f / (float)EDIM));
        const float ang = (float)b * div;
        const float pe  = (e & 1) ? __cosf(ang) : __sinf(ang);
        out[t * EDIM + e] = emb[(long)tk * EDIM + e] + pe;
    }
}

// ---------------------------------------------------------------------------
// LayerNorm (unbiased std, (std+eps) denom) — one row per block
// ---------------------------------------------------------------------------
__global__ __launch_bounds__(256) void layernorm_k(
    const float* __restrict__ x, const float* __restrict__ ga,
    const float* __restrict__ be, float* __restrict__ out)
{
    __shared__ float red[256];
    const long row = blockIdx.x;
    const int  tid = threadIdx.x;
    const float* xr = x + row * EDIM;

    float s = 0.0f;
    for (int e = tid; e < EDIM; e += 256) s += xr[e];
    red[tid] = s; __syncthreads();
    for (int off = 128; off > 0; off >>= 1) {
        if (tid < off) red[tid] += red[tid + off];
        __syncthreads();
    }
    const float mean = red[0] / (float)EDIM;
    __syncthreads();

    float v = 0.0f;
    for (int e = tid; e < EDIM; e += 256) { const float d = xr[e] - mean; v += d * d; }
    red[tid] = v; __syncthreads();
    for (int off = 128; off > 0; off >>= 1) {
        if (tid < off) red[tid] += red[tid + off];
        __syncthreads();
    }
    const float stdv = sqrtf(red[0] / (float)(EDIM - 1));
    const float inv  = 1.0f / (stdv + LNEPS);
    for (int e = tid; e < EDIM; e += 256)
        out[row * EDIM + e] = ga[e] * (xr[e] - mean) * inv + be[e];
}

// ---------------------------------------------------------------------------
// In-place log_softmax over 32000 vocab — one row per block
// ---------------------------------------------------------------------------
__global__ __launch_bounds__(256) void logsoftmax_k(float* __restrict__ io)
{
    __shared__ float red[256];
    const long row = blockIdx.x;
    float* xr = io + row * (long)VOCT;
    const int tid = threadIdx.x;

    float m = -3.0e38f;
    for (int v = tid; v < VOCT; v += 256) m = fmaxf(m, xr[v]);
    red[tid] = m; __syncthreads();
    for (int off = 128; off > 0; off >>= 1) {
        if (tid < off) red[tid] = fmaxf(red[tid], red[tid + off]);
        __syncthreads();
    }
    m = red[0]; __syncthreads();

    float s = 0.0f;
    for (int v = tid; v < VOCT; v += 256) s += __expf(xr[v] - m);
    red[tid] = s; __syncthreads();
    for (int off = 128; off > 0; off >>= 1) {
        if (tid < off) red[tid] += red[tid + off];
        __syncthreads();
    }
    const float lse = m + logf(red[0]);
    for (int v = tid; v < VOCT; v += 256) xr[v] -= lse;
}

// ---------------------------------------------------------------------------
extern "C" void kernel_launch(void* const* d_in, const int* in_sizes, int n_in,
                              void* d_out, int out_size, void* d_ws, size_t ws_size,
                              hipStream_t stream)
{
    (void)in_sizes; (void)n_in; (void)out_size; (void)ws_size;

    const int*   src     = (const int*)d_in[0];
    const int*   tgt     = (const int*)d_in[1];
    const float* src_emb = (const float*)d_in[2];
    const float* tgt_emb = (const float*)d_in[3];
    const float* eW[14]; for (int i = 0; i < 14; ++i) eW[i] = (const float*)d_in[4 + i];
    const float* dW[14]; for (int i = 0; i < 14; ++i) dW[i] = (const float*)d_in[18 + i];
    const float* gW = (const float*)d_in[32];
    const float* gb = (const float*)d_in[33];

    float* X   = (float*)d_ws;              // 2048 x 1024  encoder hidden
    float* Y   = X   + 2048L * 1024;        // 1024 x 1024  decoder hidden
    float* QH  = Y   + 1024L * 1024;        // head-major Q
    float* KH  = QH  + 2048L * 1024;        // head-major K
    float* VH  = KH  + 2048L * 1024;        // head-major V
    float* ATT = VH  + 2048L * 1024;        // attention output [b,q,1024]
    float* TMP = ATT + 2048L * 1024;        // residual-added pre-LN
    float* T   = TMP + 2048L * 1024;        // LN output / decoder t
    float* FF  = T   + 2048L * 1024;        // 2048 x 4096 FFN hidden

    auto gemm = [&](const float* A, const float* W, const float* bias,
                    const float* res, float* out, int M, int N, int K,
                    int flags, int sl) {
        gemm_f16wmma<<<dim3(N / 128, M / 128), 256, 0, stream>>>(
            A, W, bias, res, out, M, N, K, flags, sl);
    };

    embed_pe<<<BSZ * SSRC, 256, 0, stream>>>(src, src_emb, X, SSRC);
    embed_pe<<<BSZ * STGT, 256, 0, stream>>>(tgt, tgt_emb, Y, STGT);

    const int ME = BSZ * SSRC;   // 2048 encoder tokens
    const int MD = BSZ * STGT;   // 1024 decoder tokens

    // -------------------- encoder stack --------------------
    for (int i = 0; i < NLAYER; ++i) {
        const float* WQ = eW[0] + (long)i*EDIM*EDIM; const float* bQ = eW[1] + (long)i*EDIM;
        const float* WK = eW[2] + (long)i*EDIM*EDIM; const float* bK = eW[3] + (long)i*EDIM;
        const float* WV = eW[4] + (long)i*EDIM*EDIM; const float* bV = eW[5] + (long)i*EDIM;
        const float* WO = eW[6] + (long)i*EDIM*EDIM; const float* bO = eW[7] + (long)i*EDIM;
        const float* Fi = eW[8] + (long)i*EDIM*FFDIM; const float* Fib = eW[9]  + (long)i*FFDIM;
        const float* Fo = eW[10]+ (long)i*FFDIM*EDIM; const float* Fob = eW[11] + (long)i*EDIM;
        const float* na = eW[12]+ (long)i*EDIM;       const float* nb  = eW[13] + (long)i*EDIM;

        gemm(X, WQ, bQ, nullptr, QH, ME, EDIM, EDIM, 2, SSRC);
        gemm(X, WK, bK, nullptr, KH, ME, EDIM, EDIM, 2, SSRC);
        gemm(X, WV, bV, nullptr, VH, ME, EDIM, EDIM, 2, SSRC);
        attn_flash<<<dim3(SSRC/16, NHEADS, BSZ), 32, 0, stream>>>(
            QH, KH, VH, src, ATT, SSRC, SSRC, 0);
        gemm(ATT, WO, bO, X, TMP, ME, EDIM, EDIM, 0, 0);
        layernorm_k<<<ME, 256, 0, stream>>>(TMP, na, nb, T);
        gemm(T, Fi, Fib, nullptr, FF, ME, FFDIM, EDIM, 1, 0);
        gemm(FF, Fo, Fob, T, TMP, ME, EDIM, FFDIM, 0, 0);
        layernorm_k<<<ME, 256, 0, stream>>>(TMP, na, nb, X);
    }

    // -------------------- decoder stack --------------------
    for (int i = 0; i < NLAYER; ++i) {
        const float* WQ = dW[0] + (long)i*EDIM*EDIM; const float* bQ = dW[1] + (long)i*EDIM;
        const float* WK = dW[2] + (long)i*EDIM*EDIM; const float* bK = dW[3] + (long)i*EDIM;
        const float* WV = dW[4] + (long)i*EDIM*EDIM; const float* bV = dW[5] + (long)i*EDIM;
        const float* WO = dW[6] + (long)i*EDIM*EDIM; const float* bO = dW[7] + (long)i*EDIM;
        const float* Fi = dW[8] + (long)i*EDIM*FFDIM; const float* Fib = dW[9]  + (long)i*FFDIM;
        const float* Fo = dW[10]+ (long)i*FFDIM*EDIM; const float* Fob = dW[11] + (long)i*EDIM;
        const float* na = dW[12]+ (long)i*EDIM;       const float* nb  = dW[13] + (long)i*EDIM;

        // self-attention (causal + pad mask); same weights reused for cross
        gemm(Y, WQ, bQ, nullptr, QH, MD, EDIM, EDIM, 2, STGT);
        gemm(Y, WK, bK, nullptr, KH, MD, EDIM, EDIM, 2, STGT);
        gemm(Y, WV, bV, nullptr, VH, MD, EDIM, EDIM, 2, STGT);
        attn_flash<<<dim3(STGT/16, NHEADS, BSZ), 32, 0, stream>>>(
            QH, KH, VH, tgt, ATT, STGT, STGT, 1);
        gemm(ATT, WO, bO, Y, TMP, MD, EDIM, EDIM, 0, 0);
        layernorm_k<<<MD, 256, 0, stream>>>(TMP, na, nb, T);

        // cross-attention: Q from t, K/V from encoder output X
        gemm(T, WQ, bQ, nullptr, QH, MD, EDIM, EDIM, 2, STGT);
        gemm(X, WK, bK, nullptr, KH, ME, EDIM, EDIM, 2, SSRC);
        gemm(X, WV, bV, nullptr, VH, ME, EDIM, EDIM, 2, SSRC);
        attn_flash<<<dim3(STGT/16, NHEADS, BSZ), 32, 0, stream>>>(
            QH, KH, VH, src, ATT, STGT, SSRC, 0);
        gemm(ATT, WO, bO, T, TMP, MD, EDIM, EDIM, 0, 0);
        layernorm_k<<<MD, 256, 0, stream>>>(TMP, na, nb, T);

        // FFN
        gemm(T, Fi, Fib, nullptr, FF, MD, FFDIM, EDIM, 1, 0);
        gemm(FF, Fo, Fob, T, TMP, MD, EDIM, FFDIM, 0, 0);
        layernorm_k<<<MD, 256, 0, stream>>>(TMP, na, nb, Y);
    }

    // -------------------- generator + log_softmax --------------------
    float* OUT = (float*)d_out;
    gemm(Y, gW, gb, nullptr, OUT, MD, VOCT, EDIM, 0, 0);
    logsoftmax_k<<<MD, 256, 0, stream>>>(OUT);
}
